// _RPN_34093450395672
// MI455X (gfx1250) — matrix operand order repaired and checked by reading, not compiled, not run on hardware
//
#include <hip/hip_runtime.h>
#include <stdint.h>

// ---------------------------------------------------------------- types
typedef __bf16 bf16_t;
typedef __attribute__((ext_vector_type(16))) __bf16 v16bf;
typedef __attribute__((ext_vector_type(8)))  __bf16 v8bf;
typedef __attribute__((ext_vector_type(8)))  float  v8f;

union V16 { v16bf v; v8bf h[2]; };
union Pack16 { v8bf h[2]; bf16_t e[16]; };

// ---------------------------------------------------------------- consts
#define BATCH   2
#define CIN     128
#define COUT    512
#define DD      8
#define HH      32
#define WW      32
#define MPOS    (DD*HH*WW)          // 8192 positions per batch
#define SPAT    (DD*HH*WW)
#define KTOT    (CIN*27)            // 3456 (tap-major: k' = tap*128 + ci)
#define NANCH   9
#define NROIS   (MPOS*NANCH)        // 73728
#define NHEAD   80                  // valid head outputs (18 cls + 54 bbox + pad)
#define NHPAD   128                 // head weight rows padded for uniform async staging
#define KHEAD   512
#define PRE_NMS 2048
#define POST_NMS 300
#define NMS_TH  0.7f
#define LDA     40                  // LDS row stride (bf16) -> 80 B, 16B-aligned

// ---------------------------------------------------------------- async-copy helpers (CDNA5)
__device__ __forceinline__ unsigned ldsAddrOf(const void* p) {
    // LDS aperture lives in bits [63:32] of the flat address; low 32 bits = LDS offset.
    return (unsigned)(uintptr_t)p;
}
__device__ __forceinline__ void asyncLds32B(unsigned lds, unsigned gOff, unsigned long long base) {
    // two B128 async copies: 32 bytes global -> LDS (GVS mode: SGPR base + VGPR offset)
    asm volatile("global_load_async_to_lds_b128 %0, %1, %2"
                 :: "v"(lds), "v"(gOff), "s"(base) : "memory");
    asm volatile("global_load_async_to_lds_b128 %0, %1, %2 offset:16"
                 :: "v"(lds), "v"(gOff), "s"(base) : "memory");
}
template <int N>
__device__ __forceinline__ void waitAsyncLe() {
    asm volatile("s_wait_asynccnt %0" :: "i"(N) : "memory");
}

// ---------------------------------------------------------------- pack kernels
// base_feat [b][ci][z][y][x] f32 -> channels-last bf16 [b][s][ci]  (s = (z*HH+y)*WW+x)
__global__ void k_feat_cl(const float* __restrict__ src, bf16_t* __restrict__ dst) {
    int i = blockIdx.x * blockDim.x + threadIdx.x;
    if (i >= BATCH * SPAT * CIN) return;
    int ci = i & (CIN - 1);
    int t  = i >> 7;              // b*SPAT + s
    int b  = t >> 13;             // / 8192
    int s  = t & (SPAT - 1);
    dst[i] = (bf16_t)src[((size_t)b * CIN + ci) * SPAT + s];
}

// W_conv (512 x 3456 row-major, k = ci*27+tap) -> wPack[n][k'] with k' = tap*128+ci
__global__ void k_pack_convw(const float* __restrict__ w, bf16_t* __restrict__ dst) {
    int i = blockIdx.x * blockDim.x + threadIdx.x;
    if (i >= COUT * KTOT) return;
    int n  = i / KTOT;
    int kp = i - n * KTOT;
    int tap = kp >> 7;
    int ci  = kp & 127;
    dst[i] = (bf16_t)w[(size_t)n * KTOT + ci * 27 + tap];
}

// combined head weights wHeadPack[n][k] (128 x 512): n<18 W_cls, 18<=n<72 W_bbox, else 0
__global__ void k_pack_headw(const float* __restrict__ wc, const float* __restrict__ wb,
                             bf16_t* __restrict__ dst) {
    int i = blockIdx.x * blockDim.x + threadIdx.x;
    if (i >= NHPAD * KHEAD) return;
    int n = i / KHEAD, k = i - n * KHEAD;
    float v = 0.f;
    if (n < 18)      v = wc[n * KHEAD + k];
    else if (n < 72) v = wb[(n - 18) * KHEAD + k];
    dst[i] = (bf16_t)v;
}

// ---------------------------------------------------------------- conv implicit GEMM (WMMA bf16)
// grid: (N/128, M/128, B)  block: 256 (8 waves, wave32); double-buffered async staging
__global__ __launch_bounds__(256)
void k_conv_gemm(const bf16_t* __restrict__ featCl, const bf16_t* __restrict__ wPack,
                 const float* __restrict__ bias, bf16_t* __restrict__ xOut) {
    const int b     = blockIdx.z;
    const int mBase = blockIdx.y * 128;
    const int nBase = blockIdx.x * 128;

    __shared__ __align__(16) bf16_t ldsA[2][128 * LDA];
    __shared__ __align__(16) bf16_t ldsB[2][128 * LDA];

    const int tid  = threadIdx.x;
    const int lane = tid & 31;
    const int wave = tid >> 5;
    const int half = lane >> 4;
    const int r    = lane & 15;

    v8f acc[8];
    const v8f vzero = {0.f,0.f,0.f,0.f,0.f,0.f,0.f,0.f};
#pragma unroll
    for (int f = 0; f < 8; ++f) acc[f] = vzero;

    // loader coordinates: each thread stages one 32-byte row segment of A and B
    const int ldRow = tid >> 1;          // 0..127
    const int ldSeg = (tid & 1) * 16;    // 0 / 16
    const int m     = mBase + ldRow;     // spatial index s == m
    const int dz = m >> 10, hy = (m >> 5) & 31, wx = m & 31;

    const unsigned long long aBase =
        (unsigned long long)(uintptr_t)featCl + (size_t)b * SPAT * CIN * sizeof(bf16_t);
    const unsigned long long wBase = (unsigned long long)(uintptr_t)wPack;
    const unsigned aThr = (unsigned)((m * CIN + ldSeg) * sizeof(bf16_t));
    const unsigned bThr = (unsigned)(((size_t)(nBase + ldRow) * KTOT + ldSeg) * sizeof(bf16_t));

    unsigned aLds[2], bLds[2];
    aLds[0] = ldsAddrOf(&ldsA[0][ldRow * LDA + ldSeg]);
    aLds[1] = ldsAddrOf(&ldsA[1][ldRow * LDA + ldSeg]);
    bLds[0] = ldsAddrOf(&ldsB[0][ldRow * LDA + ldSeg]);
    bLds[1] = ldsAddrOf(&ldsB[1][ldRow * LDA + ldSeg]);

    // tap geometry: k0 uniform -> SALU; per-thread just 3 adds + bounds test
    auto tapInfo = [&](int k0, int& gByteOff, bool& inb) {
        int tap = k0 >> 7;
        int kd = tap / 9;
        int r9 = tap - kd * 9;
        int kh = r9 / 3;
        int kw = r9 - kh * 3;
        int zz = dz + kd - 1, yy = hy + kh - 1, xx = wx + kw - 1;
        inb = ((unsigned)zz < DD) & ((unsigned)yy < HH) & ((unsigned)xx < WW);
        int du = (kd - 1) * (HH * WW) + (kh - 1) * WW + (kw - 1);     // uniform
        gByteOff = (int)aThr + (du * CIN + (k0 & 127)) * (int)sizeof(bf16_t);
    };

    auto stage = [&](int k0, int buf) {
        int gA; bool inb;
        tapInfo(k0, gA, inb);
        unsigned gOffA = inb ? (unsigned)gA : 0u;     // clamp OOB lanes to a safe address
        asyncLds32B(aLds[buf], gOffA, aBase);
        asyncLds32B(bLds[buf], bThr + (unsigned)(k0 * sizeof(bf16_t)), wBase);
    };

    // overwrite OOB lanes with zeros (buf is a literal at each call site)
    auto zerofix = [&](int k0, int buf) {
        int gA; bool inb;
        tapInfo(k0, gA, inb);
        if (!inb) {
            Pack16 t;
#pragma unroll
            for (int j = 0; j < 16; ++j) t.e[j] = (bf16_t)0.0f;
            *(v8bf*)&ldsA[buf][ldRow * LDA + ldSeg]     = t.h[0];
            *(v8bf*)&ldsA[buf][ldRow * LDA + ldSeg + 8] = t.h[1];
        }
    };

    const int aOff = (wave * 16 + r) * LDA + half * 8;
    const int bOff = r * LDA + half * 8;

    // consume one K-step from buffer `buf` (literal index -> true DS loads)
    auto computeStep = [&](int buf) {
        V16 afrag;
        const bf16_t* pa = &ldsA[buf][aOff];
        afrag.h[0] = *(const v8bf*)(pa);
        afrag.h[1] = *(const v8bf*)(pa + 16);
#pragma unroll
        for (int f = 0; f < 8; ++f) {
            V16 bfrag;
            const bf16_t* pb = &ldsB[buf][bOff + f * 16 * LDA];
            bfrag.h[0] = *(const v8bf*)(pb);
            bfrag.h[1] = *(const v8bf*)(pb + 16);
            acc[f] = __builtin_amdgcn_wmma_f32_16x16x32_bf16(
                false, afrag.v, false, bfrag.v, (short)0, acc[f], false, false);
        }
    };

    const int NSTEP = KTOT / 32;      // 108 (even)
    stage(0, 0);
    for (int step = 0; step < NSTEP; step += 2) {
        // ---- even phase: consume buf0, prefetch step+1 -> buf1 ----
        stage((step + 1) * 32, 1);
        waitAsyncLe<4>();             // in-order completion => buf0 resident
        zerofix(step * 32, 0);
        __syncthreads();              // staged data visible to all waves
        computeStep(0);
        __syncthreads();              // consumers of buf0 done before reuse
        // ---- odd phase: consume buf1, prefetch step+2 -> buf0 ----
        if (step + 2 < NSTEP) {
            stage((step + 2) * 32, 0);
            waitAsyncLe<4>();
        } else {
            waitAsyncLe<0>();
        }
        zerofix((step + 1) * 32, 1);
        __syncthreads();
        computeStep(1);
        __syncthreads();
    }

    // ---- epilogue: bias + relu, store bf16 x[b][m][n] ----
    const int mRow = mBase + wave * 16 + half * 8;
#pragma unroll
    for (int f = 0; f < 8; ++f) {
        int n = nBase + f * 16 + r;
        float bn = bias[n];
#pragma unroll
        for (int vr = 0; vr < 8; ++vr) {
            float v = acc[f][vr] + bn;
            v = v > 0.f ? v : 0.f;
            xOut[((size_t)b * MPOS + (mRow + vr)) * COUT + n] = (bf16_t)v;
        }
    }
}

// ---------------------------------------------------------------- head GEMM (M x 512 x 80, WMMA)
// grid: (1, M/128, B) block 256; double-buffered async staging
__global__ __launch_bounds__(256)
void k_head_gemm(const bf16_t* __restrict__ xIn, const bf16_t* __restrict__ wHead,
                 const float* __restrict__ bCls, const float* __restrict__ bBbox,
                 float* __restrict__ headOut) {
    const int b     = blockIdx.z;
    const int mBase = blockIdx.y * 128;

    __shared__ __align__(16) bf16_t ldsA[2][128 * LDA];
    __shared__ __align__(16) bf16_t ldsB[2][128 * LDA];

    const int tid  = threadIdx.x;
    const int lane = tid & 31;
    const int wave = tid >> 5;
    const int half = lane >> 4;
    const int r    = lane & 15;

    v8f acc[5];
    const v8f vzero = {0.f,0.f,0.f,0.f,0.f,0.f,0.f,0.f};
#pragma unroll
    for (int f = 0; f < 5; ++f) acc[f] = vzero;

    const int ldRow = tid >> 1;
    const int ldSeg = (tid & 1) * 16;

    const unsigned long long xBase = (unsigned long long)(uintptr_t)xIn;
    const unsigned long long wBase = (unsigned long long)(uintptr_t)wHead;
    const unsigned aThr = (unsigned)((((size_t)b * MPOS + mBase + ldRow) * COUT + ldSeg) * sizeof(bf16_t));
    const unsigned bThr = (unsigned)(((size_t)ldRow * KHEAD + ldSeg) * sizeof(bf16_t));  // rows 0..127 (padded)

    unsigned aLds[2], bLds[2];
    aLds[0] = ldsAddrOf(&ldsA[0][ldRow * LDA + ldSeg]);
    aLds[1] = ldsAddrOf(&ldsA[1][ldRow * LDA + ldSeg]);
    bLds[0] = ldsAddrOf(&ldsB[0][ldRow * LDA + ldSeg]);
    bLds[1] = ldsAddrOf(&ldsB[1][ldRow * LDA + ldSeg]);

    auto stage = [&](int k0, int buf) {
        asyncLds32B(aLds[buf], aThr + (unsigned)(k0 * sizeof(bf16_t)), xBase);
        asyncLds32B(bLds[buf], bThr + (unsigned)(k0 * sizeof(bf16_t)), wBase);
    };

    const int aOff = (wave * 16 + r) * LDA + half * 8;
    const int bOff = r * LDA + half * 8;

    auto computeStep = [&](int buf) {
        V16 afrag;
        const bf16_t* pa = &ldsA[buf][aOff];
        afrag.h[0] = *(const v8bf*)(pa);
        afrag.h[1] = *(const v8bf*)(pa + 16);
#pragma unroll
        for (int f = 0; f < 5; ++f) {
            V16 bfrag;
            const bf16_t* pb = &ldsB[buf][bOff + f * 16 * LDA];
            bfrag.h[0] = *(const v8bf*)(pb);
            bfrag.h[1] = *(const v8bf*)(pb + 16);
            acc[f] = __builtin_amdgcn_wmma_f32_16x16x32_bf16(
                false, afrag.v, false, bfrag.v, (short)0, acc[f], false, false);
        }
    };

    const int NSTEP = KHEAD / 32;     // 16 (even)
    stage(0, 0);
    for (int step = 0; step < NSTEP; step += 2) {
        stage((step + 1) * 32, 1);
        waitAsyncLe<4>();
        __syncthreads();
        computeStep(0);
        __syncthreads();
        if (step + 2 < NSTEP) {
            stage((step + 2) * 32, 0);
            waitAsyncLe<4>();
        } else {
            waitAsyncLe<0>();
        }
        __syncthreads();
        computeStep(1);
        __syncthreads();
    }

    const int mRow = mBase + wave * 16 + half * 8;
#pragma unroll
    for (int f = 0; f < 5; ++f) {
        int n = f * 16 + r;
        float bn = (n < 18) ? bCls[n] : ((n < 72) ? bBbox[n - 18] : 0.f);
#pragma unroll
        for (int vr = 0; vr < 8; ++vr)
            headOut[((size_t)b * MPOS + (mRow + vr)) * NHEAD + n] = acc[f][vr] + bn;
    }
}

// ---------------------------------------------------------------- scores + box decode + clip
__global__ void k_boxes(const float* __restrict__ headOut, const float* __restrict__ imInfo,
                        float* __restrict__ boxes, float* __restrict__ scores) {
    int i = blockIdx.x * blockDim.x + threadIdx.x;
    if (i >= BATCH * NROIS) return;
    int b  = i / NROIS;
    int ii = i - b * NROIS;
    int m  = ii / NANCH;
    int a  = ii - m * NANCH;

    const float* h = headOut + ((size_t)b * MPOS + m) * NHEAD;
    float c0 = h[a], c1 = h[9 + a];
    float mx = fmaxf(c0, c1);
    float e0 = __expf(c0 - mx), e1 = __expf(c1 - mx);
    scores[i] = e1 / (e0 + e1);

    const float* dl = h + 18 + a * 6;
    float dx = dl[0], dy = dl[1], dz = dl[2], dw = dl[3], dh = dl[4], dd = dl[5];

    const float sc[3] = {4.f, 8.f, 16.f};
    const float rt[3] = {0.5f, 1.f, 2.f};
    int ridx = a / 3, sidx = a - ridx * 3;
    float aw = 8.f * sc[sidx];
    float ah = aw;
    float ad = aw * rt[ridx];
    int wx = m & 31, hy = (m >> 5) & 31, dzp = m >> 10;
    float cx = wx * 8.f + 4.f, cy = hy * 8.f + 4.f, cz = dzp * 8.f + 4.f;

    float pcx = dx * aw + cx, pcy = dy * ah + cy, pcz = dz * ad + cz;
    float pw = __expf(dw) * aw, ph = __expf(dh) * ah, pd = __expf(dd) * ad;

    const float* im = imInfo + b * 3;
    float hx = im[2] - 1.f, hyl = im[1] - 1.f, hz = im[0] - 1.f;
    float x1 = fminf(fmaxf(pcx - 0.5f * pw, 0.f), hx);
    float y1 = fminf(fmaxf(pcy - 0.5f * ph, 0.f), hyl);
    float z1 = fminf(fmaxf(pcz - 0.5f * pd, 0.f), hz);
    float x2 = fminf(fmaxf(pcx + 0.5f * pw, 0.f), hx);
    float y2 = fminf(fmaxf(pcy + 0.5f * ph, 0.f), hyl);
    float z2 = fminf(fmaxf(pcz + 0.5f * pd, 0.f), hz);

    float* bo = boxes + (size_t)i * 6;
    bo[0] = x1; bo[1] = y1; bo[2] = z1; bo[3] = x2; bo[4] = y2; bo[5] = z2;
}

// ---------------------------------------------------------------- top-k 2048 (radix select + bitonic)
__device__ __forceinline__ unsigned sortKey(float f) {
    unsigned u = __float_as_uint(f);
    return (u & 0x80000000u) ? ~u : (u | 0x80000000u);
}

__global__ __launch_bounds__(1024)
void k_topk(const float* __restrict__ scores, const float* __restrict__ boxes,
            float* __restrict__ topBoxes, int* __restrict__ topIdx) {
    const int b   = blockIdx.x;
    const int tid = threadIdx.x;
    const float* s = scores + (size_t)b * NROIS;

    __shared__ unsigned hist[256];
    __shared__ unsigned thrBin, cnt;
    __shared__ unsigned long long keys[PRE_NMS];

    for (int i = tid; i < 256; i += 1024) hist[i] = 0;
    if (tid == 0) cnt = 0;
    __syncthreads();

    for (int i = tid; i < NROIS; i += 1024)
        atomicAdd(&hist[sortKey(s[i]) >> 24], 1u);
    __syncthreads();

    if (tid == 0) {
        unsigned cum = 0; int bin = 255;
        for (; bin >= 0; --bin) { cum += hist[bin]; if (cum >= PRE_NMS) break; }
        thrBin = (bin < 0) ? 0u : (unsigned)bin;
    }
    __syncthreads();
    const unsigned tb = thrBin;

    for (int i = tid; i < NROIS; i += 1024) {
        unsigned k = sortKey(s[i]);
        if ((k >> 24) >= tb) {
            unsigned pos = atomicAdd(&cnt, 1u);
            if (pos < PRE_NMS)
                keys[pos] = ((unsigned long long)k << 32) | (unsigned)i;
        }
    }
    __syncthreads();
    unsigned c = cnt; if (c > PRE_NMS) c = PRE_NMS;
    for (int i = tid; i < PRE_NMS; i += 1024)
        if ((unsigned)i >= c) keys[i] = 0ull;

    for (unsigned kk = 2; kk <= PRE_NMS; kk <<= 1) {
        for (unsigned j = kk >> 1; j > 0; j >>= 1) {
            __syncthreads();
            for (unsigned idx = tid; idx < PRE_NMS; idx += 1024) {
                unsigned ixj = idx ^ j;
                if (ixj > idx) {
                    unsigned long long a = keys[idx], d = keys[ixj];
                    bool up = ((idx & kk) == 0);
                    bool sw = up ? (a < d) : (a > d);   // descending overall
                    if (sw) { keys[idx] = d; keys[ixj] = a; }
                }
            }
        }
    }
    __syncthreads();

    for (int i = tid; i < PRE_NMS; i += 1024) {
        unsigned idx = (unsigned)(keys[i] & 0xffffffffu);
        topIdx[b * PRE_NMS + i] = (int)idx;
        const float* src = boxes + ((size_t)b * NROIS + idx) * 6;
        float* dst = topBoxes + ((size_t)b * PRE_NMS + i) * 6;
#pragma unroll
        for (int q = 0; q < 6; ++q) dst[q] = src[q];
    }
}

// ---------------------------------------------------------------- greedy NMS + output assembly
__global__ __launch_bounds__(1024)
void k_nms(const float* __restrict__ topBoxes, float* __restrict__ out) {
    const int b   = blockIdx.x;
    const int tid = threadIdx.x;

    __shared__ float bx[PRE_NMS * 6];
    __shared__ float vol[PRE_NMS];
    __shared__ unsigned char keep[PRE_NMS];

    for (int i = tid; i < PRE_NMS; i += 1024) {
        const float* src = topBoxes + ((size_t)b * PRE_NMS + i) * 6;
        float x1 = src[0], y1 = src[1], z1 = src[2], x2 = src[3], y2 = src[4], z2 = src[5];
        float* d = bx + i * 6;
        d[0] = x1; d[1] = y1; d[2] = z1; d[3] = x2; d[4] = y2; d[5] = z2;
        vol[i]  = (x2 - x1 + 1.f) * (y2 - y1 + 1.f) * (z2 - z1 + 1.f);
        keep[i] = 1;
    }
    __syncthreads();

    for (int i = 0; i < PRE_NMS - 1; ++i) {
        if (keep[i]) {
            const float* a = bx + i * 6;
            float ax1 = a[0], ay1 = a[1], az1 = a[2], ax2 = a[3], ay2 = a[4], az2 = a[5];
            float va = vol[i];
            for (int j = i + 1 + tid; j < PRE_NMS; j += 1024) {
                if (!keep[j]) continue;
                const float* bb = bx + j * 6;
                float lx = fmaxf(ax1, bb[0]), ly = fmaxf(ay1, bb[1]), lz = fmaxf(az1, bb[2]);
                float hx = fminf(ax2, bb[3]), hy = fminf(ay2, bb[4]), hz = fminf(az2, bb[5]);
                float iw = fmaxf(hx - lx + 1.f, 0.f);
                float ih = fmaxf(hy - ly + 1.f, 0.f);
                float id = fmaxf(hz - lz + 1.f, 0.f);
                float inter = iw * ih * id;
                float iou = inter / (va + vol[j] - inter);
                if (iou > NMS_TH) keep[j] = 0;
            }
        }
        __syncthreads();
    }

    float* ob = out + (size_t)b * POST_NMS * 7;
    for (int i = tid; i < POST_NMS * 7; i += 1024) ob[i] = 0.f;
    __syncthreads();
    if (tid == 0) {
        int rank = 0;
        for (int i = 0; i < PRE_NMS && rank < POST_NMS; ++i) {
            if (keep[i]) {
                float* d = ob + rank * 7;
                d[0] = (float)b;
#pragma unroll
                for (int q = 0; q < 6; ++q) d[1 + q] = bx[i * 6 + q];
                ++rank;
            }
        }
    }
}

// ---------------------------------------------------------------- launch
extern "C" void kernel_launch(void* const* d_in, const int* in_sizes, int n_in,
                              void* d_out, int out_size, void* d_ws, size_t ws_size,
                              hipStream_t stream) {
    (void)in_sizes; (void)n_in; (void)out_size; (void)ws_size;
    const float* base_feat = (const float*)d_in[0];
    const float* im_info   = (const float*)d_in[1];
    const float* W_conv    = (const float*)d_in[4];
    const float* b_conv    = (const float*)d_in[5];
    const float* W_cls     = (const float*)d_in[6];
    const float* b_cls     = (const float*)d_in[7];
    const float* W_bbox    = (const float*)d_in[8];
    const float* b_bbox    = (const float*)d_in[9];
    float* out = (float*)d_out;

    char* ws = (char*)d_ws;
    auto carve = [&](size_t bytes) {
        char* p = ws;
        ws += (bytes + 255) & ~(size_t)255;
        return p;
    };
    bf16_t* featCl  = (bf16_t*)carve((size_t)BATCH * SPAT * CIN * 2);
    bf16_t* wConvBf = (bf16_t*)carve((size_t)COUT * KTOT * 2);
    bf16_t* wHeadBf = (bf16_t*)carve((size_t)NHPAD * KHEAD * 2);
    bf16_t* xBf     = (bf16_t*)carve((size_t)BATCH * MPOS * COUT * 2);
    float*  headF   = (float*) carve((size_t)BATCH * MPOS * NHEAD * 4);
    float*  boxesF  = (float*) carve((size_t)BATCH * NROIS * 6 * 4);
    float*  scoresF = (float*) carve((size_t)BATCH * NROIS * 4);
    float*  topBoxF = (float*) carve((size_t)BATCH * PRE_NMS * 6 * 4);
    int*    topIdx  = (int*)   carve((size_t)BATCH * PRE_NMS * 4);

    {
        int n = BATCH * SPAT * CIN;
        k_feat_cl<<<(n + 255) / 256, 256, 0, stream>>>(base_feat, featCl);
    }
    {
        int n = COUT * KTOT;
        k_pack_convw<<<(n + 255) / 256, 256, 0, stream>>>(W_conv, wConvBf);
    }
    {
        int n = NHPAD * KHEAD;
        k_pack_headw<<<(n + 255) / 256, 256, 0, stream>>>(W_cls, W_bbox, wHeadBf);
    }

    k_conv_gemm<<<dim3(COUT / 128, MPOS / 128, BATCH), 256, 0, stream>>>(
        featCl, wConvBf, b_conv, xBf);

    k_head_gemm<<<dim3(1, MPOS / 128, BATCH), 256, 0, stream>>>(
        xBf, wHeadBf, b_cls, b_bbox, headF);

    {
        int n = BATCH * NROIS;
        k_boxes<<<(n + 255) / 256, 256, 0, stream>>>(headF, im_info, boxesF, scoresF);
    }

    k_topk<<<BATCH, 1024, 0, stream>>>(scoresF, boxesF, topBoxF, topIdx);
    k_nms<<<BATCH, 1024, 0, stream>>>(topBoxF, out);
}
